// FullOp_31430570672473
// MI455X (gfx1250) — matrix operand order repaired and checked by reading, not compiled, not run on hardware
//
#include <hip/hip_runtime.h>
#include <hip/hip_bf16.h>

// MI455X (gfx1250) hypernetwork conv, bf16x3 split-precision WMMA:
//   1) hyper_gemm:  Wgen = z @ w_lin^T  (memory-bound 151MB stream) -> bf16 hi/lo [b][p][oc][ic]
//   2) pad_split:   x -> zero-padded 34x34 bf16 hi/lo
//   3) conv_wmma:   per-batch implicit GEMM, double-buffered LDS,
//                   A-tiles via global_load_async_to_lds_b128 (ASYNCcnt),
//                   B-tiles register-prefetched + ds_store_b128, v_wmma_f32_16x16x32_bf16.

typedef __attribute__((ext_vector_type(8)))  __bf16 v8bf;
typedef __attribute__((ext_vector_type(16))) __bf16 v16bf;
typedef __attribute__((ext_vector_type(8)))  float  v8f;

#define ICn   256
#define OCn   256
#define NPIX  1024
#define PADW  34
#define NB    16

static __device__ __forceinline__ unsigned short bf16_trunc(float x) {
    return (unsigned short)(__float_as_uint(x) >> 16);
}
static __device__ __forceinline__ float bf16_to_f32(unsigned short h) {
    return __uint_as_float(((unsigned)h) << 16);
}

// ---------------------------------------------------------------------------
// Kernel 1: hypernet GEMM (memory-bound).
// ---------------------------------------------------------------------------
__global__ __launch_bounds__(256) void hyper_gemm(
    const float* __restrict__ zin, const float* __restrict__ wl,
    unsigned short* __restrict__ Whi, unsigned short* __restrict__ Wlo)
{
    __shared__ float zs[16 * 64];
    __shared__ float ws[64 * 65];
    const int t = threadIdx.x;
    const int fbase = blockIdx.x * 64;

    for (int i = t; i < 1024; i += 256) zs[i] = zin[i];

    const float4* src = (const float4*)(wl + (size_t)fbase * 64);
#pragma unroll
    for (int u = 0; u < 4; ++u) {
        int g = u * 256 + t;
        float4 v = src[g];
        int flat = g * 4;
        int fi = flat >> 6, j = flat & 63;
        float* d = &ws[fi * 65 + j];
        d[0] = v.x; d[1] = v.y; d[2] = v.z; d[3] = v.w;
    }
    __syncthreads();

    const int fi = t & 63;
    const int bq = t >> 6;
    float acc[4] = {0.f, 0.f, 0.f, 0.f};
#pragma unroll 4
    for (int j = 0; j < 64; ++j) {
        float wv = ws[fi * 65 + j];
#pragma unroll
        for (int c = 0; c < 4; ++c)
            acc[c] = fmaf(zs[(bq * 4 + c) * 64 + j], wv, acc[c]);
    }

    const int f   = fbase + fi;
    const int oc  = f / 2304;
    const int rem = f - oc * 2304;
    const int ic  = rem / 9;
    const int p   = rem - ic * 9;
#pragma unroll
    for (int c = 0; c < 4; ++c) {
        int bb = bq * 4 + c;
        size_t idx = ((((size_t)bb * 9 + p) * OCn + oc) << 8) + ic;
        float v  = acc[c];
        unsigned short h = bf16_trunc(v);
        float lo = v - bf16_to_f32(h);
        Whi[idx] = h;
        Wlo[idx] = bf16_trunc(lo);
    }
}

// ---------------------------------------------------------------------------
// Kernel 2: zero-padded 34x34 bf16 hi/lo copy of x.
// ---------------------------------------------------------------------------
__global__ __launch_bounds__(256) void pad_split(
    const float* __restrict__ x,
    unsigned short* __restrict__ Xhi, unsigned short* __restrict__ Xlo, int total)
{
    int idx = blockIdx.x * 256 + threadIdx.x;
    if (idx >= total) return;
    int ww = idx % PADW; int r = idx / PADW;
    int hh = r % PADW;   r /= PADW;
    int ic = r % ICn;    int b = r / ICn;
    float v = 0.f;
    if (ww >= 1 && ww <= 32 && hh >= 1 && hh <= 32)
        v = x[(((size_t)b * ICn + ic) * 32 + (hh - 1)) * 32 + (ww - 1)];
    unsigned short h = bf16_trunc(v);
    Xhi[idx] = h;
    Xlo[idx] = bf16_trunc(v - bf16_to_f32(h));
}

// ---------------------------------------------------------------------------
// Kernel 3: conv implicit GEMM, double-buffered, async A staging.
// ---------------------------------------------------------------------------
static __device__ __forceinline__ v16bf frag_from(const unsigned short* p0,
                                                  const unsigned short* p1)
{
    v8bf a = *(const v8bf*)p0;
    v8bf c = *(const v8bf*)p1;
    v16bf r;
#pragma unroll
    for (int i = 0; i < 8; ++i) { r[i] = a[i]; r[i + 8] = c[i]; }
    return r;
}

static __device__ __forceinline__ void gatherB(
    const unsigned short* __restrict__ Xhi, const unsigned short* __restrict__ Xlo,
    size_t base0, unsigned hv[8], unsigned lv[8])
{
    const size_t icstr = (size_t)PADW * PADW;
#pragma unroll
    for (int q = 0; q < 8; ++q) {
        size_t o0 = base0 + (size_t)(2 * q) * icstr;
        size_t o1 = o0 + icstr;
        hv[q] = (unsigned)Xhi[o0] | ((unsigned)Xhi[o1] << 16);
        lv[q] = (unsigned)Xlo[o0] | ((unsigned)Xlo[o1] << 16);
    }
}

__global__ __launch_bounds__(256) void conv_wmma(
    const unsigned short* __restrict__ Whi, const unsigned short* __restrict__ Wlo,
    const unsigned short* __restrict__ Xhi, const unsigned short* __restrict__ Xlo,
    float* __restrict__ out)
{
    __shared__ __align__(16) unsigned short As[2][2][128][40]; // [buf][hi/lo][oc][ic]
    __shared__ __align__(16) unsigned short Bs[2][2][128][40]; // [buf][hi/lo][n ][ic]

    const int nblk   = blockIdx.x;   // 0..7
    const int ocblk  = blockIdx.y;   // 0..1
    const int b      = blockIdx.z;   // 0..15
    const int nbase  = nblk * 128;
    const int ocbase = ocblk * 128;
    const int h0     = nbase >> 5;

    const int t    = threadIdx.x;
    const int lane = t & 31;
    const int wid  = t >> 5;
    const int wm   = wid >> 1;
    const int wn   = wid & 1;
    const int half = lane >> 4;
    const int m    = lane & 15;

    // B staging mapping
    const int nn  = t & 127;
    const int icg = t >> 7;
    const int hq  = nn >> 5;
    const int wq  = nn & 31;

    // A async staging mapping: chunks c = t and t+256; row = c>>2, 16B seg = c&3
    const int rowA0 = t >> 2;
    const int sgA0  = (t & 3) * 16;          // bytes within 64B row payload
    const int rowA1 = (t + 256) >> 2;
    const int sgA1  = ((t + 256) & 3) * 16;

    const size_t wrowbase = (((size_t)b * 9) * OCn + ocbase) << 8; // + p*OCn<<8 later

    v8f zf = {0.f, 0.f, 0.f, 0.f, 0.f, 0.f, 0.f, 0.f};
    v8f acc[2][4];
#pragma unroll
    for (int i = 0; i < 2; ++i)
#pragma unroll
        for (int j = 0; j < 4; ++j) acc[i][j] = zf;

    // ---- helpers ----
    auto issueA = [&](int buf, int p, int ic0) {
        const unsigned short* baseH = Whi + ((((size_t)(b * 9 + p) * OCn + ocbase)) << 8);
        const unsigned short* baseL = Wlo + ((((size_t)(b * 9 + p) * OCn + ocbase)) << 8);
        unsigned g0 = (unsigned)(rowA0 * 512 + ic0 * 2 + sgA0);
        unsigned g1 = (unsigned)(rowA1 * 512 + ic0 * 2 + sgA1);
        unsigned dH0 = (unsigned)(size_t)&As[buf][0][rowA0][0] + (unsigned)sgA0;
        unsigned dH1 = (unsigned)(size_t)&As[buf][0][rowA1][0] + (unsigned)sgA1;
        unsigned dL0 = (unsigned)(size_t)&As[buf][1][rowA0][0] + (unsigned)sgA0;
        unsigned dL1 = (unsigned)(size_t)&As[buf][1][rowA1][0] + (unsigned)sgA1;
        asm volatile("global_load_async_to_lds_b128 %0, %1, %2 offset:0"
                     :: "v"(dH0), "v"(g0), "s"(baseH) : "memory");
        asm volatile("global_load_async_to_lds_b128 %0, %1, %2 offset:0"
                     :: "v"(dH1), "v"(g1), "s"(baseH) : "memory");
        asm volatile("global_load_async_to_lds_b128 %0, %1, %2 offset:0"
                     :: "v"(dL0), "v"(g0), "s"(baseL) : "memory");
        asm volatile("global_load_async_to_lds_b128 %0, %1, %2 offset:0"
                     :: "v"(dL1), "v"(g1), "s"(baseL) : "memory");
    };

    auto baseB = [&](int p, int ic0) -> size_t {
        const int kh = p / 3;
        const int kw = p - kh * 3;
        return (((size_t)b * ICn + (ic0 + icg * 16)) * PADW + (h0 + hq + kh)) * PADW
               + (wq + kw);
    };

    auto writeB = [&](int buf, const unsigned hv[8], const unsigned lv[8]) {
        uint4 u0, u1, w0, w1;
        u0.x = hv[0]; u0.y = hv[1]; u0.z = hv[2]; u0.w = hv[3];
        u1.x = hv[4]; u1.y = hv[5]; u1.z = hv[6]; u1.w = hv[7];
        w0.x = lv[0]; w0.y = lv[1]; w0.z = lv[2]; w0.w = lv[3];
        w1.x = lv[4]; w1.y = lv[5]; w1.z = lv[6]; w1.w = lv[7];
        *(uint4*)&Bs[buf][0][nn][icg * 16]     = u0;
        *(uint4*)&Bs[buf][0][nn][icg * 16 + 8] = u1;
        *(uint4*)&Bs[buf][1][nn][icg * 16]     = w0;
        *(uint4*)&Bs[buf][1][nn][icg * 16 + 8] = w1;
    };

    // ---- prologue: stage step 0 into buffer 0 ----
    {
        issueA(0, 0, 0);
        unsigned hv[8], lv[8];
        gatherB(Xhi, Xlo, baseB(0, 0), hv, lv);
        asm volatile("s_wait_asynccnt 0x0" ::: "memory");
        writeB(0, hv, lv);
        __syncthreads();
    }

    // ---- main pipelined loop: 72 K-steps (p-major, ic chunks of 32) ----
    for (int s = 0; s < 72; ++s) {
        const int buf  = s & 1;
        const bool more = (s + 1) < 72;
        unsigned hv[8], lv[8];
        if (more) {
            int sn  = s + 1;
            int pn  = sn >> 3;
            int icn = (sn & 7) * 32;
            issueA(buf ^ 1, pn, icn);               // async A(k+1)
            gatherB(Xhi, Xlo, baseB(pn, icn), hv, lv); // B(k+1) into regs
        }

        // compute(k) from LDS buf
        v16bf ah[2], al[2];
#pragma unroll
        for (int ms = 0; ms < 2; ++ms) {
            int row = wm * 32 + ms * 16 + m;
            ah[ms] = frag_from(&As[buf][0][row][8 * half], &As[buf][0][row][16 + 8 * half]);
            al[ms] = frag_from(&As[buf][1][row][8 * half], &As[buf][1][row][16 + 8 * half]);
        }
        v16bf bh[4], bl[4];
#pragma unroll
        for (int ns = 0; ns < 4; ++ns) {
            int col = wn * 64 + ns * 16 + m;
            bh[ns] = frag_from(&Bs[buf][0][col][8 * half], &Bs[buf][0][col][16 + 8 * half]);
            bl[ns] = frag_from(&Bs[buf][1][col][8 * half], &Bs[buf][1][col][16 + 8 * half]);
        }
#pragma unroll
        for (int ms = 0; ms < 2; ++ms) {
#pragma unroll
            for (int ns = 0; ns < 4; ++ns) {
                v8f c = acc[ms][ns];
                c = __builtin_amdgcn_wmma_f32_16x16x32_bf16(
                        false, ah[ms], false, bh[ns], (short)0, c, false, false);
                c = __builtin_amdgcn_wmma_f32_16x16x32_bf16(
                        false, ah[ms], false, bl[ns], (short)0, c, false, false);
                c = __builtin_amdgcn_wmma_f32_16x16x32_bf16(
                        false, al[ms], false, bh[ns], (short)0, c, false, false);
                acc[ms][ns] = c;
            }
        }

        if (more) {
            asm volatile("s_wait_asynccnt 0x0" ::: "memory"); // A(k+1) landed
            writeB(buf ^ 1, hv, lv);                          // commit B(k+1)
            __syncthreads();                                  // publish buffer k+1
        }
    }

    // ---- epilogue: D layout VGPR r -> M=r+8*half, N=m ----
#pragma unroll
    for (int ms = 0; ms < 2; ++ms) {
#pragma unroll
        for (int ns = 0; ns < 4; ++ns) {
            int nloc = wn * 64 + ns * 16 + m;
            size_t obase = ((size_t)b * OCn + ocbase + wm * 32 + ms * 16 + 8 * half)
                           * (size_t)NPIX + nbase + nloc;
#pragma unroll
            for (int r = 0; r < 8; ++r)
                out[obase + (size_t)r * NPIX] = acc[ms][ns][r];
        }
    }
}

// ---------------------------------------------------------------------------
extern "C" void kernel_launch(void* const* d_in, const int* in_sizes, int n_in,
                              void* d_out, int out_size, void* d_ws, size_t ws_size,
                              hipStream_t stream)
{
    const float* x  = (const float*)d_in[0];
    const float* z  = (const float*)d_in[1];
    const float* wl = (const float*)d_in[2];
    float* out = (float*)d_out;

    const size_t nW = (size_t)NB * 9 * OCn * ICn;     // 9,437,184
    const size_t nX = (size_t)NB * ICn * PADW * PADW; // 4,734,976
    unsigned short* Whi = (unsigned short*)d_ws;
    unsigned short* Wlo = Whi + nW;
    unsigned short* Xhi = Wlo + nW;
    unsigned short* Xlo = Xhi + nX;

    hyper_gemm<<<9216, 256, 0, stream>>>(z, wl, Whi, Wlo);

    const int totalPad = (int)nX;
    pad_split<<<(totalPad + 255) / 256, 256, 0, stream>>>(x, Xhi, Xlo, totalPad);

    conv_wmma<<<dim3(8, 2, NB), 256, 0, stream>>>(Whi, Wlo, Xhi, Xlo, out);
}
